// CasualSelfAttention_10943576670586
// MI455X (gfx1250) — compile-verified, matrix operand order checked by reading, not run on hardware
//
#include <hip/hip_runtime.h>

// ---------------------------------------------------------------------------
// Causal self-attention (GQA + RoPE) for MI455X / gfx1250.
// All matmuls run on v_wmma_f32_16x16x32_bf16 (bf16 in, f32 accumulate).
// B = 1, S = 2048, D = 4096, NH = 32, NKV = 8, HD = 128.
// Weights are pre-transposed during bf16 conversion so every LDS stage is a
// contiguous b128 copy; GEMM is templated on N/K so all strides fold into
// immediate offsets.
// ---------------------------------------------------------------------------

static constexpr int SLEN = 2048;
static constexpr int DMODEL = 4096;
static constexpr int NHEADS = 32;
static constexpr int NKVH = 8;
static constexpr int HDIM = 128;

typedef __bf16 bf16;
typedef __attribute__((ext_vector_type(16))) bf16 v16bf;
typedef __attribute__((ext_vector_type(8)))  bf16 v8bf;
typedef __attribute__((ext_vector_type(8)))  float v8f;

static __device__ inline v8f vzero8f() {
    v8f z;
#pragma unroll
    for (int i = 0; i < 8; ++i) z[i] = 0.0f;
    return z;
}

// Build a 16-element bf16 fragment from two contiguous 8-element (16B) chunks.
// Matches the CDNA5 16-bit A/B fragment layout: each lane holds two
// contiguous 8-element runs of the K dimension.
static __device__ inline v16bf mk_frag(const bf16* lo_p, const bf16* hi_p) {
    v8bf lo = *(const v8bf*)lo_p;
    v8bf hi = *(const v8bf*)hi_p;
    return __builtin_shufflevector(lo, hi, 0, 1, 2, 3, 4, 5, 6, 7,
                                           8, 9, 10, 11, 12, 13, 14, 15);
}

static __device__ inline v8f wmma_bf16(v16bf a, v16bf b, v8f c) {
    // (neg_a, A, neg_b, B, c_mod, C, reuse_a, reuse_b)
    return __builtin_amdgcn_wmma_f32_16x16x32_bf16(false, a, false, b,
                                                   (short)0, c, false, false);
}

// ---------------------------------------------------------------------------
// f32 -> bf16 conversion (grid-stride), plain layout
// ---------------------------------------------------------------------------
__global__ void cvt_f32_bf16(const float* __restrict__ in,
                             bf16* __restrict__ out, int n) {
    int i = blockIdx.x * blockDim.x + threadIdx.x;
    int stride = gridDim.x * blockDim.x;
    for (; i < n; i += stride) out[i] = (bf16)in[i];
}

// f32 [R][Cc] -> bf16 [Cc][R]  (transpose weights once so GEMM B operand is
// row-major over K and stages into LDS with contiguous 16B copies)
__global__ void cvt_f32_bf16_t(const float* __restrict__ in,
                               bf16* __restrict__ out, int R, int Cc) {
    int i = blockIdx.x * blockDim.x + threadIdx.x;
    int stride = gridDim.x * blockDim.x;
    int n = R * Cc;
    for (; i < n; i += stride) {
        int r = i / Cc;
        int c = i - r * Cc;
        out[(size_t)c * R + r] = (bf16)in[i];
    }
}

// ---------------------------------------------------------------------------
// Tiled bf16 GEMM: C[f32, MxN] = A[bf16, MxK row-major] * Bt[bf16, NxK]^T
// Block: 128x256 tile, 256 threads = 8 waves (2x4), each wave 64x64
// (4x4 fragments -> 16 WMMA per 32-wide k-step from 8 fragment loads).
// Software-pipelined: next global tile loads overlap current WMMA burst.
// M % 128 == 0, N % 256 == 0, K % 32 == 0 for all uses here.
// ---------------------------------------------------------------------------
template <int N, int K>
__global__ __launch_bounds__(256)
void gemm_bf16_tn(const bf16* __restrict__ A,    // [M][K]
                  const bf16* __restrict__ Bt,   // [N][K]
                  float* __restrict__ C) {       // [M][N]
    __shared__ __align__(16) bf16 As[128 * 32];  // [row][k]
    __shared__ __align__(16) bf16 Bs[256 * 32];  // [col][k]

    const int tid  = threadIdx.x;
    const int w    = tid >> 5;
    const int lane = tid & 31;
    const int lm   = lane & 15;
    const int kb   = (lane < 16) ? 0 : 8;        // per-lane K chunk base
    const int mh   = (lane >> 4) * 8;            // C-layout row offset
    const int M0   = blockIdx.y * 128;
    const int N0   = blockIdx.x * 256;
    const int wm   = w >> 2;                     // 0..1 -> 64-row strip
    const int wn   = w & 3;                      // 0..3 -> 64-col strip

    // Per-thread staging coordinates (contiguous 16B chunks)
    const int ar[2] = {(tid + 0) >> 2, (tid + 256) >> 2};         // 0..127
    const int ac    = (tid & 3) * 8;                              // 0..24
    int br[4];
#pragma unroll
    for (int p = 0; p < 4; ++p) br[p] = (tid + p * 256) >> 2;     // 0..255

    v8f acc[4][4];
#pragma unroll
    for (int mi = 0; mi < 4; ++mi)
#pragma unroll
        for (int ni = 0; ni < 4; ++ni) acc[mi][ni] = vzero8f();

    // Prologue: fetch k-step 0 into registers
    v8bf aR[2], bR[4];
#pragma unroll
    for (int p = 0; p < 2; ++p)
        aR[p] = *(const v8bf*)(A + (size_t)(M0 + ar[p]) * K + ac);
#pragma unroll
    for (int p = 0; p < 4; ++p)
        bR[p] = *(const v8bf*)(Bt + (size_t)(N0 + br[p]) * K + ac);

    for (int k0 = 0; k0 < K; k0 += 32) {
        __syncthreads();                  // previous iteration done with LDS
        // Commit staged registers to LDS
#pragma unroll
        for (int p = 0; p < 2; ++p)
            *(v8bf*)&As[ar[p] * 32 + ac] = aR[p];
#pragma unroll
        for (int p = 0; p < 4; ++p)
            *(v8bf*)&Bs[br[p] * 32 + ac] = bR[p];
        // Kick off next tile's global loads (overlap with WMMA below)
        if (k0 + 32 < K) {
            const int kn = k0 + 32;
#pragma unroll
            for (int p = 0; p < 2; ++p)
                aR[p] = *(const v8bf*)(A + (size_t)(M0 + ar[p]) * K + kn + ac);
#pragma unroll
            for (int p = 0; p < 4; ++p)
                bR[p] = *(const v8bf*)(Bt + (size_t)(N0 + br[p]) * K + kn + ac);
            // Pull the tile after next toward L2 (global_prefetch_b8)
            if (k0 + 64 < K) {
                __builtin_prefetch(A + (size_t)(M0 + ar[0]) * K + kn + 32 + ac, 0, 1);
                __builtin_prefetch(Bt + (size_t)(N0 + br[0]) * K + kn + 32 + ac, 0, 1);
            }
        }
        __syncthreads();                  // LDS tile ready

        v16bf af[4], bfr[4];
#pragma unroll
        for (int mi = 0; mi < 4; ++mi) {
            int row = wm * 64 + mi * 16 + lm;
            af[mi] = mk_frag(&As[row * 32 + kb], &As[row * 32 + kb + 16]);
        }
#pragma unroll
        for (int ni = 0; ni < 4; ++ni) {
            int col = wn * 64 + ni * 16 + lm;
            bfr[ni] = mk_frag(&Bs[col * 32 + kb], &Bs[col * 32 + kb + 16]);
        }
#pragma unroll
        for (int mi = 0; mi < 4; ++mi)
#pragma unroll
            for (int ni = 0; ni < 4; ++ni)
                acc[mi][ni] = wmma_bf16(af[mi], bfr[ni], acc[mi][ni]);
    }

    // Epilogue: one base pointer per lane; all strides are compile-time so
    // every store gets an immediate offset (max (63*N + 48)*4 < 8MB).
    float* cb = C + (size_t)(M0 + wm * 64 + mh) * N + N0 + wn * 64 + lm;
#pragma unroll
    for (int mi = 0; mi < 4; ++mi)
#pragma unroll
        for (int ni = 0; ni < 4; ++ni)
#pragma unroll
            for (int v = 0; v < 8; ++v)
                cb[(size_t)(mi * 16 + v) * N + ni * 16] = acc[mi][ni][v];
}

// ---------------------------------------------------------------------------
// RoPE (interleaved pair) + layout change + bf16 convert.
// X: [S][nheads*HD] f32  ->  Y: [nheads][S][HD] bf16
// ---------------------------------------------------------------------------
__global__ void rope_f32_to_bf16(const float* __restrict__ X,
                                 bf16* __restrict__ Y, int nheads) {
    int idx = blockIdx.x * blockDim.x + threadIdx.x;
    int total = SLEN * nheads * (HDIM / 2);
    if (idx >= total) return;
    int i = idx & 63;          // pair index 0..63
    int t = idx >> 6;
    int head = t % nheads;
    int s = t / nheads;
    size_t bi = (size_t)s * (nheads * HDIM) + head * HDIM + 2 * i;
    float x0 = X[bi], x1 = X[bi + 1];
    // inv_freq = 500000^(-(2i)/128); ln(500000) = 13.1223633774...
    float freq = __expf(-((float)(2 * i) * (1.0f / 128.0f)) * 13.122363377404328f);
    float ang = (float)s * freq;
    float c, sn;
    __sincosf(ang, &c, &sn);
    size_t bo = ((size_t)head * SLEN + s) * HDIM + 2 * i;
    Y[bo]     = (bf16)(x0 * c - x1 * sn);
    Y[bo + 1] = (bf16)(x0 * sn + x1 * c);
}

// V: [S][NKV*HD] f32 -> Vt: [NKV][HD][S] bf16 (so PV B-tiles stage contiguously)
__global__ void v_transpose_bf16(const float* __restrict__ Vf,
                                 bf16* __restrict__ Vt) {
    int idx = blockIdx.x * blockDim.x + threadIdx.x;
    int total = SLEN * NKVH * HDIM;
    if (idx >= total) return;
    int d = idx & 127;
    int t = idx >> 7;
    int hh = t % NKVH;
    int s = t / NKVH;
    Vt[((size_t)hh * HDIM + d) * SLEN + s] =
        (bf16)Vf[(size_t)s * (NKVH * HDIM) + hh * HDIM + d];
}

// ---------------------------------------------------------------------------
// Flash attention (causal, GQA). One block per (head, 128-row q block).
// 256 threads = 8 waves; wave w owns 16 q rows. LDS: one shared 128x128 bf16
// tile (K then V, barrier-separated) + per-wave 16x128 bf16 P scratch = 64KB.
// ---------------------------------------------------------------------------
__global__ __launch_bounds__(256)
void attn_kernel(const bf16* __restrict__ Qb,   // [NH][S][HD]
                 const bf16* __restrict__ Kb,   // [NKV][S][HD]
                 const bf16* __restrict__ Vt,   // [NKV][HD][S]
                 bf16* __restrict__ Ao) {       // [S][NH*HD]
    __shared__ __align__(16) bf16 tile[128 * 128];       // K block, then V block
    __shared__ __align__(16) bf16 Plds[8][16 * 128];     // per-wave P scratch

    const int tid  = threadIdx.x;
    const int w    = tid >> 5;
    const int lane = tid & 31;
    const int lm   = lane & 15;
    const int half = lane >> 4;
    const int kb   = half ? 8 : 0;
    const int h    = blockIdx.x >> 4;     // q head
    const int qb   = blockIdx.x & 15;     // q block (128 rows each)
    const int hkv  = h >> 2;              // GQA: 4 q heads per kv head

    // Q fragments for this wave's 16 rows (kept in registers for all key blocks)
    const bf16* qp =
        Qb + ((size_t)h * SLEN + (size_t)qb * 128 + w * 16 + lm) * HDIM;
    v16bf qf[4];
#pragma unroll
    for (int kk = 0; kk < 4; ++kk)
        qf[kk] = mk_frag(qp + kk * 32 + kb, qp + kk * 32 + kb + 16);

    v8f oacc[8];
#pragma unroll
    for (int nt = 0; nt < 8; ++nt) oacc[nt] = vzero8f();
    float mrun[8], lrun[8];
#pragma unroll
    for (int v = 0; v < 8; ++v) { mrun[v] = -3.0e38f; lrun[v] = 0.0f; }

    const float scale = 0.08838834764831845f;   // 1/sqrt(128)

    for (int j = 0; j <= qb; ++j) {
        __syncthreads();
        // Stage K block: tile[key][hd]
#pragma unroll
        for (int p = 0; p < 8; ++p) {
            int t = tid + p * 256;
            int r = t >> 4;
            int c = (t & 15) * 8;
            *(v8bf*)&tile[r * 128 + c] = *(const v8bf*)(
                Kb + ((size_t)hkv * SLEN + (size_t)j * 128 + r) * HDIM + c);
        }
        __syncthreads();

        // S = Q * K^T  (B operand: column n = key, K dim = hd)
        v8f sacc[8];
#pragma unroll
        for (int nt = 0; nt < 8; ++nt) sacc[nt] = vzero8f();
#pragma unroll
        for (int nt = 0; nt < 8; ++nt) {
            int key = nt * 16 + lm;
#pragma unroll
            for (int kk = 0; kk < 4; ++kk) {
                v16bf bfr = mk_frag(&tile[key * 128 + kk * 32 + kb],
                                    &tile[key * 128 + kk * 32 + kb + 16]);
                sacc[nt] = wmma_bf16(qf[kk], bfr, sacc[nt]);
            }
        }
        __syncthreads();   // everyone done reading the K tile

        // Stage V block: tile[hd][key]
#pragma unroll
        for (int p = 0; p < 8; ++p) {
            int t = tid + p * 256;
            int r = t >> 4;
            int c = (t & 15) * 8;
            *(v8bf*)&tile[r * 128 + c] = *(const v8bf*)(
                Vt + ((size_t)hkv * HDIM + r) * SLEN + (size_t)j * 128 + c);
        }

        // Scale + causal mask (C layout: m = v + 8*half, n = nt*16 + lm)
        const bool diag = (j == qb);
#pragma unroll
        for (int nt = 0; nt < 8; ++nt)
#pragma unroll
            for (int v = 0; v < 8; ++v) {
                float s = sacc[nt][v] * scale;
                if (diag) {
                    int m_in = w * 16 + v + 8 * half;
                    int n_in = nt * 16 + lm;
                    if (n_in > m_in) s = -1.0e30f;
                }
                sacc[nt][v] = s;
            }

        // Online softmax per row slot; reductions stay within each 16-lane half
        bf16* Pw = &Plds[w][0];
#pragma unroll
        for (int v = 0; v < 8; ++v) {
            float pm = sacc[0][v];
#pragma unroll
            for (int nt = 1; nt < 8; ++nt) pm = fmaxf(pm, sacc[nt][v]);
            for (int off = 1; off < 16; off <<= 1)
                pm = fmaxf(pm, __shfl_xor(pm, off, 32));
            float mnew = fmaxf(mrun[v], pm);
            float fac = __expf(mrun[v] - mnew);
            float rsum = 0.0f;
#pragma unroll
            for (int nt = 0; nt < 8; ++nt) {
                float pe = __expf(sacc[nt][v] - mnew);
                sacc[nt][v] = pe;
                rsum += pe;
            }
            for (int off = 1; off < 16; off <<= 1)
                rsum += __shfl_xor(rsum, off, 32);
            lrun[v] = lrun[v] * fac + rsum;
            mrun[v] = mnew;
#pragma unroll
            for (int nt = 0; nt < 8; ++nt) oacc[nt][v] *= fac;
            // Stash P row (C layout -> row-major LDS) for A-fragment reload
            int m_in = v + 8 * half;
#pragma unroll
            for (int nt = 0; nt < 8; ++nt)
                Pw[m_in * 128 + nt * 16 + lm] = (bf16)sacc[nt][v];
        }
        __syncthreads();   // V tile staged; P visible (own-wave anyway)

        // O += P * V  (A = P 16x128 from LDS, B column n = hd, K dim = key)
        v16bf pf[4];
#pragma unroll
        for (int kk = 0; kk < 4; ++kk)
            pf[kk] = mk_frag(&Pw[lm * 128 + kk * 32 + kb],
                             &Pw[lm * 128 + kk * 32 + kb + 16]);
#pragma unroll
        for (int nt = 0; nt < 8; ++nt) {
            int hdcol = nt * 16 + lm;
#pragma unroll
            for (int kk = 0; kk < 4; ++kk) {
                v16bf bfr = mk_frag(&tile[hdcol * 128 + kk * 32 + kb],
                                    &tile[hdcol * 128 + kk * 32 + kb + 16]);
                oacc[nt] = wmma_bf16(pf[kk], bfr, oacc[nt]);
            }
        }
    }

    // Normalize and emit bf16 rows of [S][NH*HD] for the output projection
#pragma unroll
    for (int nt = 0; nt < 8; ++nt)
#pragma unroll
        for (int v = 0; v < 8; ++v) {
            int row = qb * 128 + w * 16 + v + 8 * half;
            int col = h * HDIM + nt * 16 + lm;
            Ao[(size_t)row * (NHEADS * HDIM) + col] =
                (bf16)(oacc[nt][v] / lrun[v]);
        }
}

// ---------------------------------------------------------------------------
// Host-side launch sequence
// ---------------------------------------------------------------------------
extern "C" void kernel_launch(void* const* d_in, const int* in_sizes, int n_in,
                              void* d_out, int out_size, void* d_ws,
                              size_t ws_size, hipStream_t stream) {
    const float* x  = (const float*)d_in[0];
    const float* Wq = (const float*)d_in[1];
    const float* Wk = (const float*)d_in[2];
    const float* Wv = (const float*)d_in[3];
    const float* Wo = (const float*)d_in[4];
    float* out = (float*)d_out;

    char* ws = (char*)d_ws;
    size_t off = 0;
    auto walloc = [&](size_t bytes) -> char* {
        char* p = ws + off;
        off += (bytes + 255) & ~(size_t)255;
        return p;
    };
    bf16*  xb   = (bf16*)walloc((size_t)SLEN * DMODEL * 2);
    bf16*  Wqt  = (bf16*)walloc((size_t)DMODEL * (NHEADS * HDIM) * 2); // [N][K]
    bf16*  Wkt  = (bf16*)walloc((size_t)DMODEL * (NKVH * HDIM) * 2);   // [N][K]
    bf16*  Wvt  = (bf16*)walloc((size_t)DMODEL * (NKVH * HDIM) * 2);   // [N][K]
    bf16*  Wot  = (bf16*)walloc((size_t)DMODEL * DMODEL * 2);          // [N][K]
    float* Qf   = (float*)walloc((size_t)SLEN * (NHEADS * HDIM) * 4);
    float* Kf   = (float*)walloc((size_t)SLEN * (NKVH * HDIM) * 4);
    float* Vf   = (float*)walloc((size_t)SLEN * (NKVH * HDIM) * 4);
    bf16*  Qb   = (bf16*)walloc((size_t)NHEADS * SLEN * HDIM * 2);
    bf16*  Kb   = (bf16*)walloc((size_t)NKVH * SLEN * HDIM * 2);
    bf16*  Vt   = (bf16*)walloc((size_t)NKVH * HDIM * SLEN * 2);
    bf16*  Aob  = (bf16*)walloc((size_t)SLEN * (NHEADS * HDIM) * 2);

    dim3 b256(256);

    // 1. bf16 conversions (weights transposed to [N][K])
    cvt_f32_bf16<<<2048, b256, 0, stream>>>(x, xb, SLEN * DMODEL);
    cvt_f32_bf16_t<<<4096, b256, 0, stream>>>(Wq, Wqt, DMODEL, NHEADS * HDIM);
    cvt_f32_bf16_t<<<1024, b256, 0, stream>>>(Wk, Wkt, DMODEL, NKVH * HDIM);
    cvt_f32_bf16_t<<<1024, b256, 0, stream>>>(Wv, Wvt, DMODEL, NKVH * HDIM);
    cvt_f32_bf16_t<<<4096, b256, 0, stream>>>(Wo, Wot, DMODEL, DMODEL);

    // 2. QKV projections (WMMA). Block tile 128x256.
    gemm_bf16_tn<NHEADS * HDIM, DMODEL>
        <<<dim3((NHEADS * HDIM) / 256, SLEN / 128), b256, 0, stream>>>(xb, Wqt, Qf);
    gemm_bf16_tn<NKVH * HDIM, DMODEL>
        <<<dim3((NKVH * HDIM) / 256, SLEN / 128), b256, 0, stream>>>(xb, Wkt, Kf);
    gemm_bf16_tn<NKVH * HDIM, DMODEL>
        <<<dim3((NKVH * HDIM) / 256, SLEN / 128), b256, 0, stream>>>(xb, Wvt, Vf);

    // 3. RoPE + layout/dtype shuffles
    rope_f32_to_bf16<<<(SLEN * NHEADS * 64 + 255) / 256, b256, 0, stream>>>(
        Qf, Qb, NHEADS);
    rope_f32_to_bf16<<<(SLEN * NKVH * 64 + 255) / 256, b256, 0, stream>>>(
        Kf, Kb, NKVH);
    v_transpose_bf16<<<(SLEN * NKVH * HDIM + 255) / 256, b256, 0, stream>>>(
        Vf, Vt);

    // 4. Flash attention (WMMA)
    attn_kernel<<<NHEADS * (SLEN / 128), b256, 0, stream>>>(Qb, Kb, Vt, Aob);

    // 5. Output projection (WMMA) straight into d_out
    gemm_bf16_tn<DMODEL, DMODEL>
        <<<dim3(DMODEL / 256, SLEN / 128), b256, 0, stream>>>(Aob, Wot, out);
}
